// RegionProposalNetwork_15839839387737
// MI455X (gfx1250) — compile-verified, matrix operand order checked by reading, not run on hardware
//
#include <hip/hip_runtime.h>
#include <hip/hip_bf16.h>
#include <stdint.h>

// ---------------------------------------------------------------------------
// MI455X (gfx1250) RPN implementation.
//  - conv3x3 + heads as implicit GEMM in bf16 via v_wmma_f32_16x16x32_bf16
//  - double-buffered LDS staging through global_load_async_to_lds_b128
//    (ASYNCcnt); pipeline manually unrolled x2 so the buffer index is a
//    compile-time constant (no cndmask pointer selects in the hot loop)
//  - strength-reduced addressing: A offsets linear in ks, conv taps via
//    packed 2-bit tables (no div/mod in the hot loop)
//  - sort/NMS fully LDS-resident (exploits 320KB/WGP)
// ---------------------------------------------------------------------------

typedef __attribute__((ext_vector_type(16))) __bf16 v16bf;
typedef __attribute__((ext_vector_type(8)))  float  v8f;
typedef __attribute__((ext_vector_type(4)))  unsigned int u32x4;

#define NIMG   2
#define CIN    512
#define IMH    50
#define IMW    50
#define HWPIX  2500
#define COUT   512
#define KTOT   4608      // 9 * 512
#define KSTEPS 144       // KTOT / 32
#define NANCH  9
#define NLOC   36
#define MHEAD  64        // 36 loc rows + 18 score rows + 10 pad
#define NPRE   12000
#define NPOST  2000
#define SORTN  32768
#define NBOX   22500     // HWPIX * NANCH

// packed (ky+1)/(kx+1) per tap, 2 bits each
#define KYTAB  0x2A540u
#define KXTAB  0x92424u

// d_out layout (float elements): pred_locs, pred_scores, rois, roi_indices, anchors
#define OFF_LOC   0
#define OFF_SCORE 180000
#define OFF_ROIS  270000
#define OFF_IDX   286000
#define OFF_ANCH  290000

__device__ __forceinline__ unsigned short f2bf(float f) {
  unsigned u = __float_as_uint(f);
  u += 0x7FFFu + ((u >> 16) & 1u);   // round-to-nearest-even
  return (unsigned short)(u >> 16);
}

// 16B-per-lane async copy global -> LDS (tracked by ASYNCcnt).
__device__ __forceinline__ void async_g2l_b128(const void* gp, void* lp) {
  unsigned long long ga = (unsigned long long)gp;
  unsigned la = (unsigned)(unsigned long long)lp;   // low 32 bits = LDS offset
  asm volatile("global_load_async_to_lds_b128 %0, %1, off"
               :: "v"(la), "v"(ga) : "memory");
}
__device__ __forceinline__ void wait_async() {
  asm volatile("s_wait_asynccnt 0x0" ::: "memory");
}

// ---------------------------------------------------------------------------
// Prep kernels: fp32 -> bf16 repacks into GEMM-friendly K-contiguous layouts
// ---------------------------------------------------------------------------
__global__ void k_prep_x(const float* __restrict__ x, unsigned short* __restrict__ xb) {
  int i = blockIdx.x * blockDim.x + threadIdx.x;     // NIMG*HWPIX*CIN
  if (i >= NIMG * HWPIX * CIN) return;
  int c = i % CIN; int t = i / CIN;
  int pix = t % HWPIX; int img = t / HWPIX;
  xb[i] = f2bf(x[((size_t)img * CIN + c) * HWPIX + pix]);
}

__global__ void k_prep_w(const float* __restrict__ W1, unsigned short* __restrict__ wa) {
  int i = blockIdx.x * blockDim.x + threadIdx.x;     // COUT*KTOT
  if (i >= COUT * KTOT) return;
  int k = i % KTOT; int o = i / KTOT;
  int tap = k / CIN; int ci = k % CIN;
  int ky = tap / 3, kx = tap % 3;
  wa[i] = f2bf(W1[(((size_t)o * CIN + ci) * 3 + ky) * 3 + kx]);
}

__global__ void k_prep_wh(const float* __restrict__ Wl, const float* __restrict__ Ws,
                          unsigned short* __restrict__ wh) {
  int i = blockIdx.x * blockDim.x + threadIdx.x;     // MHEAD*CIN
  if (i >= MHEAD * CIN) return;
  int c = i % CIN; int m = i / CIN;
  float v = 0.f;
  if (m < NLOC)            v = Wl[m * CIN + c];
  else if (m < NLOC + 18)  v = Ws[(m - NLOC) * CIN + c];
  wh[i] = f2bf(v);
}

// ---------------------------------------------------------------------------
// conv3x3 + bias + ReLU:  C[cout][pix] = sum_k W[cout][k] * im2col(x)[k][pix]
// block = 256 thr (8 waves), tile 128(M) x 128(N), wave tile 32x64,
// double-buffered async LDS staging, pipeline unrolled x2.
// ---------------------------------------------------------------------------
__global__ void __launch_bounds__(256)
k_conv3x3(const unsigned short* __restrict__ xb,
          const unsigned short* __restrict__ wa,
          const float* __restrict__ b1,
          unsigned short* __restrict__ h1) {
  __shared__ __align__(16) unsigned short As[2][128 * 32];
  __shared__ __align__(16) unsigned short Bs[2][128 * 32];

  const int tid  = threadIdx.x;
  const int lane = tid & 31;
  const int wv   = tid >> 5;
  const int cout0 = blockIdx.x * 128;
  const int pix0  = blockIdx.y * 128;
  const int img   = blockIdx.z;
  const int wm = wv & 3;    // 0..3 -> M offset
  const int wn = wv >> 2;   // 0..1 -> N offset

  // ---- hoisted per-thread staging state (2 x 16B chunks for A and B) ----
  int rowS[2], kkS[2];
  const unsigned short* aBase[2];          // A global base, +ks*32 each step
  bool pOK[2]; int pY[2], pX[2];
  #pragma unroll
  for (int j = 0; j < 2; ++j) {
    int c0 = tid * 2 + j;
    rowS[j] = c0 >> 2;
    kkS[j]  = (c0 & 3) * 8;
    aBase[j] = wa + (size_t)(cout0 + rowS[j]) * KTOT + kkS[j];
    int p = pix0 + rowS[j];
    pOK[j] = (p < HWPIX);
    pY[j] = p / IMW; pX[j] = p % IMW;
  }

  auto stage = [&](int ks, int buf) {
    const int tap = ks >> 4;
    const int ky = (int)((KYTAB >> (2 * tap)) & 3u) - 1;
    const int kx = (int)((KXTAB >> (2 * tap)) & 3u) - 1;
    #pragma unroll
    for (int j = 0; j < 2; ++j)
      async_g2l_b128(aBase[j] + (size_t)ks * 32, &As[buf][rowS[j] * 32 + kkS[j]]);
    #pragma unroll
    for (int j = 0; j < 2; ++j) {
      int yy = pY[j] + ky, xx = pX[j] + kx;
      bool ok = pOK[j] && ((unsigned)yy < IMH) && ((unsigned)xx < IMW);
      unsigned short* lp = &Bs[buf][rowS[j] * 32 + kkS[j]];
      if (ok) {
        async_g2l_b128(xb + ((size_t)(img * IMH + yy) * IMW + xx) * CIN
                          + (size_t)(ks & 15) * 32 + kkS[j], lp);
      } else {
        u32x4 z = {0u, 0u, 0u, 0u};
        *(u32x4*)lp = z;
      }
    }
  };

  v8f acc[2][4];
  #pragma unroll
  for (int a = 0; a < 2; ++a)
    #pragma unroll
    for (int b = 0; b < 4; ++b)
      #pragma unroll
      for (int r = 0; r < 8; ++r) acc[a][b][r] = 0.f;

  const int kg  = (lane >> 4) * 8;          // K-group per ISA 16-bit layout
  const int l15 = lane & 15;

  auto compute = [&](int buf) {
    union { u32x4 q[2]; v16bf v; } af[2], bfm[4];
    #pragma unroll
    for (int mi = 0; mi < 2; ++mi) {
      int row = wm * 32 + mi * 16 + l15;
      af[mi].q[0] = *(const u32x4*)&As[buf][row * 32 + kg];
      af[mi].q[1] = *(const u32x4*)&As[buf][row * 32 + 16 + kg];
    }
    #pragma unroll
    for (int ni = 0; ni < 4; ++ni) {
      int row = wn * 64 + ni * 16 + l15;
      bfm[ni].q[0] = *(const u32x4*)&Bs[buf][row * 32 + kg];
      bfm[ni].q[1] = *(const u32x4*)&Bs[buf][row * 32 + 16 + kg];
    }
    #pragma unroll
    for (int mi = 0; mi < 2; ++mi)
      #pragma unroll
      for (int ni = 0; ni < 4; ++ni)
        acc[mi][ni] = __builtin_amdgcn_wmma_f32_16x16x32_bf16(
            false, af[mi].v, false, bfm[ni].v, (short)0, acc[mi][ni], false, false);
  };

  stage(0, 0);                              // prologue fill
  for (int ks = 0; ks < KSTEPS; ks += 2) {  // KSTEPS even: buf parity static
    wait_async();
    __syncthreads();
    stage(ks + 1, 1);                       // overlap next DMA with WMMA
    compute(0);
    wait_async();
    __syncthreads();
    if (ks + 2 < KSTEPS) stage(ks + 2, 0);
    compute(1);
  }

  // epilogue: bias + ReLU, store bf16 NHWC for the head GEMM
  const int mo = (lane >= 16) ? 8 : 0;
  #pragma unroll
  for (int mi = 0; mi < 2; ++mi) {
    int cbase = cout0 + wm * 32 + mi * 16 + mo;
    #pragma unroll
    for (int ni = 0; ni < 4; ++ni) {
      int p = pix0 + wn * 64 + ni * 16 + l15;
      if (p < HWPIX) {
        size_t ob = ((size_t)img * HWPIX + p) * COUT;
        #pragma unroll
        for (int r = 0; r < 8; ++r) {
          float v = acc[mi][ni][r] + b1[cbase + r];
          h1[ob + cbase + r] = f2bf(v > 0.f ? v : 0.f);
        }
      }
    }
  }
}

// ---------------------------------------------------------------------------
// fused 1x1 heads: 64x512 (Wl|Ws) GEMM against h1, fp32 out, double-buffered
// ---------------------------------------------------------------------------
__global__ void __launch_bounds__(256)
k_head(const unsigned short* __restrict__ h1,
       const unsigned short* __restrict__ wh,
       float* __restrict__ hout) {
  __shared__ __align__(16) unsigned short As[2][64 * 32];
  __shared__ __align__(16) unsigned short Bs[2][128 * 32];
  const int tid = threadIdx.x, lane = tid & 31, wv = tid >> 5;
  const int pix0 = blockIdx.x * 128, img = blockIdx.y;

  // A: 1 chunk/thread, B: 2 chunks/thread; offsets linear in ks
  const int rowA = tid >> 2, kkA = (tid & 3) * 8;
  const unsigned short* aBase = wh + (size_t)rowA * CIN + kkA;

  int rowB[2], kkB[2]; bool pOK[2];
  const unsigned short* bBase[2];
  #pragma unroll
  for (int j = 0; j < 2; ++j) {
    int c0 = tid * 2 + j;
    rowB[j] = c0 >> 2; kkB[j] = (c0 & 3) * 8;
    int p = pix0 + rowB[j];
    pOK[j] = (p < HWPIX);
    bBase[j] = h1 + ((size_t)img * HWPIX + (pOK[j] ? p : 0)) * CIN + kkB[j];
  }

  auto stage = [&](int ks, int buf) {
    async_g2l_b128(aBase + (size_t)ks * 32, &As[buf][rowA * 32 + kkA]);
    #pragma unroll
    for (int j = 0; j < 2; ++j) {
      unsigned short* lp = &Bs[buf][rowB[j] * 32 + kkB[j]];
      if (pOK[j]) {
        async_g2l_b128(bBase[j] + (size_t)ks * 32, lp);
      } else {
        u32x4 z = {0u, 0u, 0u, 0u};
        *(u32x4*)lp = z;
      }
    }
  };

  v8f acc[4];
  #pragma unroll
  for (int a = 0; a < 4; ++a)
    #pragma unroll
    for (int r = 0; r < 8; ++r) acc[a][r] = 0.f;

  const int kg = (lane >> 4) * 8;
  const int l15 = lane & 15;

  auto compute = [&](int buf) {
    union { u32x4 q[2]; v16bf v; } af, bfm;
    { int row = wv * 16 + l15;
      bfm.q[0] = *(const u32x4*)&Bs[buf][row * 32 + kg];
      bfm.q[1] = *(const u32x4*)&Bs[buf][row * 32 + 16 + kg]; }
    #pragma unroll
    for (int mi = 0; mi < 4; ++mi) {
      int row = mi * 16 + l15;
      af.q[0] = *(const u32x4*)&As[buf][row * 32 + kg];
      af.q[1] = *(const u32x4*)&As[buf][row * 32 + 16 + kg];
      acc[mi] = __builtin_amdgcn_wmma_f32_16x16x32_bf16(
          false, af.v, false, bfm.v, (short)0, acc[mi], false, false);
    }
  };

  stage(0, 0);
  for (int ks = 0; ks < 16; ks += 2) {
    wait_async();
    __syncthreads();
    stage(ks + 1, 1);
    compute(0);
    wait_async();
    __syncthreads();
    if (ks + 2 < 16) stage(ks + 2, 0);
    compute(1);
  }

  const int mo = (lane >= 16) ? 8 : 0;
  const int p = pix0 + wv * 16 + l15;
  if (p < HWPIX) {
    size_t ob = ((size_t)img * HWPIX + p) * MHEAD;
    #pragma unroll
    for (int mi = 0; mi < 4; ++mi)
      #pragma unroll
      for (int r = 0; r < 8; ++r)
        hout[ob + mi * 16 + mo + r] = acc[mi][r];
  }
}

// ---------------------------------------------------------------------------
// head post: biases, pred_locs/pred_scores outputs, fg softmax
// ---------------------------------------------------------------------------
__global__ void k_head_post(const float* __restrict__ hout,
                            const float* __restrict__ bl, const float* __restrict__ bs,
                            float* __restrict__ out, float* __restrict__ fgs) {
  int i = blockIdx.x * blockDim.x + threadIdx.x;    // NIMG*HWPIX*NANCH
  if (i >= NIMG * HWPIX * NANCH) return;
  int a = i % NANCH; int t = i / NANCH;
  int pix = t % HWPIX; int img = t / HWPIX;
  const float* hb = &hout[((size_t)img * HWPIX + pix) * MHEAD];
  int row = pix * NANCH + a;
  #pragma unroll
  for (int d = 0; d < 4; ++d)
    out[OFF_LOC + ((size_t)img * NBOX + row) * 4 + d] = hb[a * 4 + d] + bl[a * 4 + d];
  float s0 = hb[NLOC + a * 2]     + bs[a * 2];
  float s1 = hb[NLOC + a * 2 + 1] + bs[a * 2 + 1];
  out[OFF_SCORE + ((size_t)img * NBOX + row) * 2]     = s0;
  out[OFF_SCORE + ((size_t)img * NBOX + row) * 2 + 1] = s1;
  fgs[(size_t)img * NBOX + row] = 1.f / (1.f + __expf(s0 - s1));
}

__global__ void k_anchors(float* __restrict__ out) {
  int i = blockIdx.x * blockDim.x + threadIdx.x;
  if (i >= NBOX) return;
  int a = i % NANCH; int pix = i / NANCH;
  float cy = (float)(pix / IMW) * 16.f + 8.f;
  float cx = (float)(pix % IMW) * 16.f + 8.f;
  const float ratios[3] = {0.5f, 1.f, 2.f};
  const float scales[3] = {8.f, 16.f, 32.f};
  float r = ratios[a / 3], s = scales[a % 3];
  float h = 16.f * s * __fsqrt_rn(r);
  float w = 16.f * s * __fsqrt_rn(1.f / r);
  out[OFF_ANCH + (size_t)i * 4 + 0] = cy - 0.5f * h;
  out[OFF_ANCH + (size_t)i * 4 + 1] = cx - 0.5f * w;
  out[OFF_ANCH + (size_t)i * 4 + 2] = cy + 0.5f * h;
  out[OFF_ANCH + (size_t)i * 4 + 3] = cx + 0.5f * w;
}

__global__ void k_decode(const float* __restrict__ out, const float* __restrict__ fgs,
                         const int* __restrict__ ihp, const int* __restrict__ iwp,
                         float* __restrict__ boxes, float* __restrict__ scs) {
  int i = blockIdx.x * blockDim.x + threadIdx.x;    // NIMG*NBOX
  if (i >= NIMG * NBOX) return;
  int j = i % NBOX;
  float a0 = out[OFF_ANCH + (size_t)j * 4 + 0];
  float a1 = out[OFF_ANCH + (size_t)j * 4 + 1];
  float a2 = out[OFF_ANCH + (size_t)j * 4 + 2];
  float a3 = out[OFF_ANCH + (size_t)j * 4 + 3];
  float ah = a2 - a0, aw = a3 - a1;
  float ay = a0 + 0.5f * ah, ax = a1 + 0.5f * aw;
  const float* L = &out[OFF_LOC + (size_t)i * 4];
  float cy = L[0] * ah + ay, cx = L[1] * aw + ax;
  float h = __expf(L[2]) * ah, w = __expf(L[3]) * aw;
  float imh = (float)ihp[0], imw = (float)iwp[0];
  float y1 = fminf(fmaxf(cy - 0.5f * h, 0.f), imh);
  float x1 = fminf(fmaxf(cx - 0.5f * w, 0.f), imw);
  float y2 = fminf(fmaxf(cy + 0.5f * h, 0.f), imh);
  float x2 = fminf(fmaxf(cx + 0.5f * w, 0.f), imw);
  boxes[(size_t)i * 4 + 0] = y1; boxes[(size_t)i * 4 + 1] = x1;
  boxes[(size_t)i * 4 + 2] = y2; boxes[(size_t)i * 4 + 3] = x2;
  scs[i] = ((y2 - y1) >= 16.f && (x2 - x1) >= 16.f) ? fgs[i] : -__builtin_inff();
}

// ---------------------------------------------------------------------------
// LDS-resident bitonic sort (32768 keys+idx = 256KB of the 320KB WGP LDS)
// ---------------------------------------------------------------------------
__global__ void __launch_bounds__(1024)
k_sort(const float* __restrict__ scs, const float* __restrict__ boxes,
       float* __restrict__ tops, float* __restrict__ topb) {
  extern __shared__ __align__(16) unsigned char smem[];
  float* key = (float*)smem;
  int*   idx = (int*)(key + SORTN);
  const int tid = threadIdx.x, img = blockIdx.x;
  for (int i = tid; i < SORTN; i += 1024) {
    key[i] = (i < NBOX) ? scs[(size_t)img * NBOX + i] : -__builtin_inff();
    idx[i] = i;
  }
  __syncthreads();
  for (int k = 2; k <= SORTN; k <<= 1) {
    for (int j = k >> 1; j > 0; j >>= 1) {
      for (int i = tid; i < SORTN; i += 1024) {
        int p = i ^ j;
        if (p > i) {
          float a = key[i], b = key[p];
          bool desc = ((i & k) == 0);           // descending overall
          if ((a < b) == desc) {
            key[i] = b; key[p] = a;
            int t = idx[i]; idx[i] = idx[p]; idx[p] = t;
          }
        }
      }
      __syncthreads();
    }
  }
  for (int i = tid; i < NPRE; i += 1024) {
    tops[(size_t)img * NPRE + i] = key[i];
    int s = idx[i];
    #pragma unroll
    for (int d = 0; d < 4; ++d)
      topb[((size_t)img * NPRE + i) * 4 + d] =
          (s < NBOX) ? boxes[((size_t)img * NBOX + s) * 4 + d] : 0.f;
  }
}

// ---------------------------------------------------------------------------
// LDS-resident greedy NMS: 12000 boxes+scores (240KB) live in LDS,
// 2000 sequential argmax+suppress rounds; writes rois and roi_indices.
// ---------------------------------------------------------------------------
__global__ void __launch_bounds__(1024)
k_nms(const float* __restrict__ tops, const float* __restrict__ topb,
      float* __restrict__ out) {
  extern __shared__ __align__(16) unsigned char smem[];
  float* bx = (float*)smem;          // NPRE*4
  float* sc = bx + NPRE * 4;         // NPRE
  __shared__ float rv[1024];
  __shared__ int   ri[1024];
  const int tid = threadIdx.x, img = blockIdx.x;

  for (int i = tid; i < NPRE; i += 1024) {
    sc[i] = tops[(size_t)img * NPRE + i];
    #pragma unroll
    for (int d = 0; d < 4; ++d)
      bx[i * 4 + d] = topb[((size_t)img * NPRE + i) * 4 + d];
  }
  __syncthreads();

  for (int it = 0; it < NPOST; ++it) {
    float bv = -__builtin_inff(); int bi = 0;
    for (int i = tid; i < NPRE; i += 1024) {
      float v = sc[i];
      if (v > bv) { bv = v; bi = i; }
    }
    rv[tid] = bv; ri[tid] = bi;
    __syncthreads();
    for (int s = 512; s > 0; s >>= 1) {
      if (tid < s && rv[tid + s] > rv[tid]) { rv[tid] = rv[tid + s]; ri[tid] = ri[tid + s]; }
      __syncthreads();
    }
    const int best = ri[0];
    const bool valid = rv[0] > -1e37f;
    const float b0 = bx[best * 4], b1 = bx[best * 4 + 1];
    const float b2 = bx[best * 4 + 2], b3 = bx[best * 4 + 3];
    const float barea = (b2 - b0) * (b3 - b1);
    if (tid < 4) out[OFF_ROIS + ((size_t)img * NPOST + it) * 4 + tid] = valid ? bx[best * 4 + tid] : 0.f;
    if (tid == 0) ((int*)out)[OFF_IDX + img * NPOST + it] = img;
    __syncthreads();
    for (int i = tid; i < NPRE; i += 1024) {
      float v = sc[i];
      if (valid && v > -1e37f) {
        float c0 = bx[i * 4], c1 = bx[i * 4 + 1], c2 = bx[i * 4 + 2], c3 = bx[i * 4 + 3];
        float iy1 = fmaxf(b0, c0), ix1 = fmaxf(b1, c1);
        float iy2 = fminf(b2, c2), ix2 = fminf(b3, c3);
        float inter = fmaxf(iy2 - iy1, 0.f) * fmaxf(ix2 - ix1, 0.f);
        float area = (c2 - c0) * (c3 - c1);
        float iou = inter / (barea + area - inter + 1e-9f);
        if (iou > 0.7f) sc[i] = -__builtin_inff();
      }
    }
    if (tid == 0) sc[best] = -__builtin_inff();
    __syncthreads();
  }
}

// ---------------------------------------------------------------------------
extern "C" void kernel_launch(void* const* d_in, const int* in_sizes, int n_in,
                              void* d_out, int out_size, void* d_ws, size_t ws_size,
                              hipStream_t stream) {
  (void)in_sizes; (void)n_in; (void)out_size; (void)ws_size;
  const float* x  = (const float*)d_in[0];
  const float* W1 = (const float*)d_in[1];
  const float* b1 = (const float*)d_in[2];
  const float* Ws = (const float*)d_in[3];
  const float* bs = (const float*)d_in[4];
  const float* Wl = (const float*)d_in[5];
  const float* bl = (const float*)d_in[6];
  const int*   ih = (const int*)d_in[7];
  const int*   iw = (const int*)d_in[8];
  float* out = (float*)d_out;

  uint8_t* ws = (uint8_t*)d_ws;
  size_t off = 0;
  auto take = [&](size_t bytes) -> void* {
    void* p = ws + off; off += (bytes + 255) & ~(size_t)255; return p;
  };
  unsigned short* xb   = (unsigned short*)take((size_t)NIMG * HWPIX * CIN * 2);
  unsigned short* wa   = (unsigned short*)take((size_t)COUT * KTOT * 2);
  unsigned short* h1   = (unsigned short*)take((size_t)NIMG * HWPIX * CIN * 2);
  unsigned short* wh   = (unsigned short*)take((size_t)MHEAD * CIN * 2);
  float*          hout = (float*)take((size_t)NIMG * HWPIX * MHEAD * 4);
  float*          fgs  = (float*)take((size_t)NIMG * NBOX * 4);
  float*          bxs  = (float*)take((size_t)NIMG * NBOX * 4 * 4);
  float*          scs  = (float*)take((size_t)NIMG * NBOX * 4);
  float*          topb = (float*)take((size_t)NIMG * NPRE * 4 * 4);
  float*          tops = (float*)take((size_t)NIMG * NPRE * 4);

  k_prep_x <<<(NIMG * HWPIX * CIN + 255) / 256, 256, 0, stream>>>(x, xb);
  k_prep_w <<<(COUT * KTOT + 255) / 256, 256, 0, stream>>>(W1, wa);
  k_prep_wh<<<(MHEAD * CIN + 255) / 256, 256, 0, stream>>>(Wl, Ws, wh);

  k_conv3x3<<<dim3(COUT / 128, 20, NIMG), 256, 0, stream>>>(xb, wa, b1, h1);
  k_head   <<<dim3(20, NIMG), 256, 0, stream>>>(h1, wh, hout);

  k_head_post<<<(NIMG * HWPIX * NANCH + 255) / 256, 256, 0, stream>>>(hout, bl, bs, out, fgs);
  k_anchors  <<<(NBOX + 255) / 256, 256, 0, stream>>>(out);
  k_decode   <<<(NIMG * NBOX + 255) / 256, 256, 0, stream>>>(out, fgs, ih, iw, bxs, scs);

  k_sort<<<NIMG, 1024, (size_t)SORTN * 8, stream>>>(scs, bxs, tops, topb);
  k_nms <<<NIMG, 1024, (size_t)NPRE * 5 * 4, stream>>>(tops, topb, out);
}
